// TransBlock_61005715472510
// MI455X (gfx1250) — matrix operand order repaired and checked by reading, not compiled, not run on hardware
//
#include <hip/hip_runtime.h>

typedef __attribute__((ext_vector_type(16))) __bf16 v16bf;
typedef __attribute__((ext_vector_type(8)))  float  v8f;

union FragB { uint4 u[2]; v16bf v; };
union Acc8  { float f[8]; v8f v; };

#define WMMA_BF16(A,B,C) \
  __builtin_amdgcn_wmma_f32_16x16x32_bf16(false,(A),false,(B),(short)0,(C),false,false)

// CDNA5 async global->LDS copy (ASYNCcnt path), with safe fallback.
#if defined(__HIP_DEVICE_COMPILE__) && \
    __has_builtin(__builtin_amdgcn_global_load_async_to_lds_b128) && \
    __has_builtin(__builtin_amdgcn_s_wait_asynccnt)
#define HAVE_ASYNC_LDS 1
typedef int v4i __attribute__((vector_size(4 * sizeof(int))));
typedef __attribute__((address_space(1))) v4i gv4i;
typedef __attribute__((address_space(3))) v4i lv4i;
#define ASYNC_CP16(dst_lds, src_glob) \
  __builtin_amdgcn_global_load_async_to_lds_b128((gv4i*)(src_glob), (lv4i*)(dst_lds), 0, 0)
#define ASYNC_WAIT() __builtin_amdgcn_s_wait_asynccnt(0)
#else
#define HAVE_ASYNC_LDS 0
#define ASYNC_CP16(dst_lds, src_glob) \
  do { *(uint4*)(dst_lds) = *(const uint4*)(src_glob); } while (0)
#define ASYNC_WAIT() do {} while (0)
#endif

// ---------------- fp32 -> bf16 convert ----------------
__global__ __launch_bounds__(256) void k_cvt_bf16(const float* __restrict__ in,
                                                  __bf16* __restrict__ out, int n) {
  int i = (blockIdx.x * 256 + threadIdx.x) * 4;
  if (i + 3 < n) {
    float4 v = *(const float4*)(in + i);
    out[i + 0] = (__bf16)v.x; out[i + 1] = (__bf16)v.y;
    out[i + 2] = (__bf16)v.z; out[i + 3] = (__bf16)v.w;
  }
}

// ---------------- W[K][N] fp32 -> Wt[N][K] bf16 ----------------
__global__ __launch_bounds__(256) void k_cvt_wt(const float* __restrict__ W,
                                                __bf16* __restrict__ Wt, int K, int N) {
  __shared__ float t[32][33];
  int nb = blockIdx.x * 32, kb = blockIdx.y * 32;
  int tx = threadIdx.x, ty = threadIdx.y;
  for (int i = ty; i < 32; i += 8)
    t[i][tx] = W[(size_t)(kb + i) * N + nb + tx];
  __syncthreads();
  for (int i = ty; i < 32; i += 8)
    Wt[(size_t)(nb + i) * K + kb + tx] = (__bf16)t[tx][i];
}

// ---------------- 128x128 tile WMMA GEMM: C = X @ Wt^T + bias ----------------
// mode 0: out_bf16[m][n] = v*scale          (Q: scale=1/8, K: scale=1)
// mode 1: out_bf16 transposed per batch ->  Vt[b][n][pos]
// mode 2: out_bf16 = gelu(v)                (FFN1)
// mode 3: out_f32  = v + res[m][n]          (FFN2 + residual)
__global__ __launch_bounds__(256) void k_gemm_wmma(
    const __bf16* __restrict__ X, const __bf16* __restrict__ Wt,
    const float* __restrict__ bias, const float* __restrict__ res,
    float* __restrict__ outF, __bf16* __restrict__ outB,
    int M, int N, int K, int mode, float scale)
{
  __shared__ __align__(16) __bf16 lb[128 * 40];   // B tile, padded row stride 80B
  const int tid = threadIdx.x, wave = tid >> 5, lane = tid & 31;
  const int lrow = lane & 15, kh = (lane >> 4) * 8, kb16 = (lane >> 4) * 16;
  const int m0 = blockIdx.y * 128 + wave * 16;
  const int n0 = blockIdx.x * 128;

  Acc8 acc[8];
#pragma unroll
  for (int t = 0; t < 8; ++t)
#pragma unroll
    for (int r = 0; r < 8; ++r) acc[t].f[r] = 0.0f;

  const int r64 = tid >> 2, c8 = (tid & 3) * 8;

  for (int k0 = 0; k0 < K; k0 += 32) {
    __syncthreads();           // everyone done reading lb from previous step
#pragma unroll
    for (int i = 0; i < 2; ++i) {
      int row = r64 + i * 64;  // async (or fallback) copy of W tile into LDS
      ASYNC_CP16(&lb[row * 40 + c8], Wt + (size_t)(n0 + row) * K + k0 + c8);
    }
    FragB a;  // A frag: row lrow, contiguous K kb16..kb16+15
    const __bf16* ap = X + (size_t)(m0 + lrow) * K + k0 + kb16;
    a.u[0] = *(const uint4*)(ap);
    a.u[1] = *(const uint4*)(ap + 8);
    ASYNC_WAIT();
    __syncthreads();
#pragma unroll
    for (int t = 0; t < 8; ++t) {
      FragB b;  // B frag: col lrow, contiguous K kb16..kb16+15
      const __bf16* bp = &lb[(t * 16 + lrow) * 40 + kb16];
      b.u[0] = *(const uint4*)(bp);
      b.u[1] = *(const uint4*)(bp + 8);
      acc[t].v = WMMA_BF16(a.v, b.v, acc[t].v);
    }
  }

#pragma unroll
  for (int t = 0; t < 8; ++t) {
    int n = n0 + t * 16 + lrow;
    float bv = bias[n];
#pragma unroll
    for (int r = 0; r < 8; ++r) {
      int m = m0 + kh + r;
      float v = (acc[t].f[r] + bv) * scale;
      if (mode == 0) {
        outB[(size_t)m * N + n] = (__bf16)v;
      } else if (mode == 1) {
        int bb = m >> 11, pos = m & 2047;
        outB[((size_t)bb * N + n) * 2048 + pos] = (__bf16)v;
      } else if (mode == 2) {
        float gl = 0.5f * v * (1.0f + erff(v * 0.70710678118f));
        outB[(size_t)m * N + n] = (__bf16)gl;
      } else {
        outF[(size_t)m * N + n] = v + res[(size_t)m * N + n];
      }
    }
  }
}

// ---------------- attention + talking heads, one q-tile (16 rows) per block ----------------
// 16 waves, wave h handles head h.  Qb pre-scaled by 1/sqrt(HD).
__global__ __launch_bounds__(512) void k_attn(
    const __bf16* __restrict__ Qb,   // [B][L][D] head-major, scaled
    const __bf16* __restrict__ Kb,   // [B][L][D]
    const __bf16* __restrict__ Vt,   // [B][D][L]
    const float* __restrict__ Wm, const float* __restrict__ bm,
    float* __restrict__ outA)        // [B][L][D]
{
  const int L = 2048, D = 1024;
  __shared__ __align__(16) __bf16 pe[512 * 40];   // [elem 16q*32k][head 0..15, 16..31 zero]
  __shared__ __align__(16) __bf16 pg[256 * 40];   // [g][m][32k]
  __shared__ __align__(16) __bf16 wmb[16 * 40];   // Wm rows, cols 16..31 zero
  __shared__ float bmsh[16];

  const int tid = threadIdx.x, h = tid >> 5, lane = tid & 31;
  const int lrow = lane & 15, kh = (lane >> 4) * 8, kb16 = (lane >> 4) * 16;
  const int b = blockIdx.x >> 7, q0 = (blockIdx.x & 127) << 4;

  { // zero pe padding/head-pad and wmb
    unsigned* p = (unsigned*)pe;
    for (int i = tid; i < 512 * 20; i += 512) p[i] = 0u;
    unsigned* w = (unsigned*)wmb;
    for (int i = tid; i < 16 * 20; i += 512) w[i] = 0u;
  }
  __syncthreads();
  if (tid < 256) wmb[(tid >> 4) * 40 + (tid & 15)] = (__bf16)Wm[tid];
  if (tid < 16) bmsh[tid] = bm[tid];

  // Q fragments (A: 16q x 32hd), kept in registers for both passes
  FragB qf0, qf1;
  const __bf16* qp = Qb + ((size_t)b * L + q0 + lrow) * D + h * 64;
  qf0.u[0] = *(const uint4*)(qp + kb16);      qf0.u[1] = *(const uint4*)(qp + kb16 + 8);
  qf1.u[0] = *(const uint4*)(qp + 32 + kb16); qf1.u[1] = *(const uint4*)(qp + 32 + kb16 + 8);

  const __bf16* kbase = Kb + (size_t)b * L * D + h * 64;

  // ---- pass A: Z[q] = sum_k exp(s) ----
  float zp[8];
#pragma unroll
  for (int r = 0; r < 8; ++r) zp[r] = 0.0f;
  for (int kt = 0; kt < L; kt += 16) {
    const __bf16* kp = kbase + (size_t)(kt + lrow) * D;
    FragB kf0, kf1;  // B frags: col = key pos, contiguous hd kb16..+15
    kf0.u[0] = *(const uint4*)(kp + kb16);      kf0.u[1] = *(const uint4*)(kp + kb16 + 8);
    kf1.u[0] = *(const uint4*)(kp + 32 + kb16); kf1.u[1] = *(const uint4*)(kp + 32 + kb16 + 8);
    Acc8 s;
#pragma unroll
    for (int r = 0; r < 8; ++r) s.f[r] = 0.0f;
    s.v = WMMA_BF16(qf0.v, kf0.v, s.v);
    s.v = WMMA_BF16(qf1.v, kf1.v, s.v);
#pragma unroll
    for (int r = 0; r < 8; ++r) zp[r] += __expf(s.f[r]);
  }
#pragma unroll
  for (int off = 8; off > 0; off >>= 1)
#pragma unroll
    for (int r = 0; r < 8; ++r) zp[r] += __shfl_xor(zp[r], off, 32);
  float rz[8];
#pragma unroll
  for (int r = 0; r < 8; ++r) rz[r] = 1.0f / zp[r];

  __syncthreads();

  // constant mixing A-frag: Wm (rows g, K = heads 0..15, zero-padded to 32)
  FragB wf;
  wf.u[0] = *(const uint4*)(&wmb[lrow * 40 + kb16]);
  wf.u[1] = *(const uint4*)(&wmb[lrow * 40 + kb16 + 8]);   // hi lanes read zero pad

  Acc8 accO[4];
#pragma unroll
  for (int t = 0; t < 4; ++t)
#pragma unroll
    for (int r = 0; r < 8; ++r) accO[t].f[r] = 0.0f;

  const __bf16* vtb = Vt + (size_t)b * D * L;

  // ---- pass B ----
  for (int kt = 0; kt < L; kt += 32) {
    Acc8 s0, s1;
#pragma unroll
    for (int r = 0; r < 8; ++r) { s0.f[r] = 0.f; s1.f[r] = 0.f; }
    {
      const __bf16* kp = kbase + (size_t)(kt + lrow) * D;
      FragB kf0, kf1;
      kf0.u[0] = *(const uint4*)(kp + kb16);      kf0.u[1] = *(const uint4*)(kp + kb16 + 8);
      kf1.u[0] = *(const uint4*)(kp + 32 + kb16); kf1.u[1] = *(const uint4*)(kp + 32 + kb16 + 8);
      s0.v = WMMA_BF16(qf0.v, kf0.v, s0.v);
      s0.v = WMMA_BF16(qf1.v, kf1.v, s0.v);
      kp += (size_t)16 * D;
      kf0.u[0] = *(const uint4*)(kp + kb16);      kf0.u[1] = *(const uint4*)(kp + kb16 + 8);
      kf1.u[0] = *(const uint4*)(kp + 32 + kb16); kf1.u[1] = *(const uint4*)(kp + 32 + kb16 + 8);
      s1.v = WMMA_BF16(qf0.v, kf0.v, s1.v);
      s1.v = WMMA_BF16(qf1.v, kf1.v, s1.v);
    }
#pragma unroll
    for (int r = 0; r < 8; ++r) {           // normalized probs -> LDS [elem][head]
      int m = kh + r;
      pe[(m * 32 + lrow) * 40 + h]      = (__bf16)(__expf(s0.f[r]) * rz[r]);
      pe[(m * 32 + 16 + lrow) * 40 + h] = (__bf16)(__expf(s1.f[r]) * rz[r]);
    }
    __syncthreads();
    // talking-heads mix via WMMA: D[g][e] = Wm @ P ; wave handles elem groups 2h, 2h+1
#pragma unroll
    for (int gi = 0; gi < 2; ++gi) {
      int e0 = (h * 2 + gi) * 16;
      FragB pf;
      const __bf16* pp = &pe[(e0 + lrow) * 40 + kb16];   // hi lanes read zero pad
      pf.u[0] = *(const uint4*)(pp);
      pf.u[1] = *(const uint4*)(pp + 8);
      Acc8 d;
#pragma unroll
      for (int r = 0; r < 8; ++r) d.f[r] = 0.f;
      d.v = WMMA_BF16(wf.v, pf.v, d.v);
      int e = e0 + lrow, m = e >> 5, kk = e & 31;
#pragma unroll
      for (int r = 0; r < 8; ++r) {
        int gp = kh + r;
        pg[(gp * 16 + m) * 40 + kk] = (__bf16)(d.f[r] + bmsh[gp]);  // +bm -> bm*colsum(V)
      }
    }
    __syncthreads();
    // P'' @ V for head g == h
    FragB af;
    const __bf16* ppg = &pg[(h * 16 + lrow) * 40 + kb16];
    af.u[0] = *(const uint4*)(ppg);
    af.u[1] = *(const uint4*)(ppg + 8);
#pragma unroll
    for (int nt = 0; nt < 4; ++nt) {
      FragB vf;
      const __bf16* vp = vtb + (size_t)(h * 64 + nt * 16 + lrow) * L + kt;
      vf.u[0] = *(const uint4*)(vp + kb16);
      vf.u[1] = *(const uint4*)(vp + kb16 + 8);
      accO[nt].v = WMMA_BF16(af.v, vf.v, accO[nt].v);
    }
    __syncthreads();
  }

  float* op = outA + ((size_t)b * L + q0) * D + h * 64;
#pragma unroll
  for (int nt = 0; nt < 4; ++nt)
#pragma unroll
    for (int r = 0; r < 8; ++r)
      op[(size_t)(kh + r) * D + nt * 16 + lrow] = accO[nt].f[r];
}

// ---------------- residual + LayerNorm, one wave per row ----------------
__global__ __launch_bounds__(256) void k_add_ln(
    const float* __restrict__ attn, const float* __restrict__ qin,
    const float* __restrict__ g, const float* __restrict__ beta,
    float* __restrict__ xf, __bf16* __restrict__ xb)
{
  const int lane = threadIdx.x & 31;
  const size_t row = (size_t)blockIdx.x * 8 + (threadIdx.x >> 5);
  const float* pa = attn + row * 1024;
  const float* pq = qin + row * 1024;
  float v[32]; float s = 0.f, s2 = 0.f;
#pragma unroll
  for (int i = 0; i < 32; ++i) {
    float x = pa[lane + i * 32] + pq[lane + i * 32];
    v[i] = x; s += x; s2 += x * x;
  }
#pragma unroll
  for (int off = 16; off > 0; off >>= 1) {
    s += __shfl_xor(s, off, 32); s2 += __shfl_xor(s2, off, 32);
  }
  float mean = s * (1.0f / 1024.0f);
  float var  = s2 * (1.0f / 1024.0f) - mean * mean;
  float rstd = rsqrtf(var + 1e-5f);
#pragma unroll
  for (int i = 0; i < 32; ++i) {
    int c = lane + i * 32;
    float y = (v[i] - mean) * rstd * g[c] + beta[c];
    xf[row * 1024 + c] = y;
    xb[row * 1024 + c] = (__bf16)y;
  }
}

extern "C" void kernel_launch(void* const* d_in, const int* in_sizes, int n_in,
                              void* d_out, int out_size, void* d_ws, size_t ws_size,
                              hipStream_t stream) {
  const float* q   = (const float*)d_in[0];
  const float* Wq  = (const float*)d_in[1];
  const float* bq  = (const float*)d_in[2];
  const float* Wk  = (const float*)d_in[3];
  const float* bk  = (const float*)d_in[4];
  const float* Wv  = (const float*)d_in[5];
  const float* bv  = (const float*)d_in[6];
  const float* Wm  = (const float*)d_in[7];
  const float* bm  = (const float*)d_in[8];
  const float* W1  = (const float*)d_in[9];
  const float* b1  = (const float*)d_in[10];
  const float* W2  = (const float*)d_in[11];
  const float* b2  = (const float*)d_in[12];
  const float* lng = (const float*)d_in[13];
  const float* lnb = (const float*)d_in[14];

  const int Mtot = 4096, Dm = 1024, L = 2048, Bb = 2;
  char* ws = (char*)d_ws;
  size_t off = 0;
  auto alloc = [&](size_t bytes) {
    char* p = ws + off; off += (bytes + 255) & ~(size_t)255; return p;
  };
  __bf16* qb   = (__bf16*)alloc((size_t)Mtot * Dm * 2);
  __bf16* wtq  = (__bf16*)alloc((size_t)Dm * Dm * 2);
  __bf16* wtk  = (__bf16*)alloc((size_t)Dm * Dm * 2);
  __bf16* wtv  = (__bf16*)alloc((size_t)Dm * Dm * 2);
  __bf16* wt1  = (__bf16*)alloc((size_t)Dm * Dm * 2);
  __bf16* wt2  = (__bf16*)alloc((size_t)Dm * Dm * 2);
  __bf16* Qbm  = (__bf16*)alloc((size_t)Mtot * Dm * 2);
  __bf16* Kbm  = (__bf16*)alloc((size_t)Mtot * Dm * 2);
  __bf16* Vtm  = (__bf16*)alloc((size_t)Mtot * Dm * 2);
  float*  attn = (float*) alloc((size_t)Mtot * Dm * 4);
  float*  xf   = (float*) alloc((size_t)Mtot * Dm * 4);
  __bf16* xb   = (__bf16*)alloc((size_t)Mtot * Dm * 2);
  __bf16* hb   = (__bf16*)alloc((size_t)Mtot * Dm * 2);

  k_cvt_bf16<<<Mtot * Dm / 1024, 256, 0, stream>>>(q, qb, Mtot * Dm);
  dim3 tb(32, 8), tg(Dm / 32, Dm / 32);
  k_cvt_wt<<<tg, tb, 0, stream>>>(Wq, wtq, Dm, Dm);
  k_cvt_wt<<<tg, tb, 0, stream>>>(Wk, wtk, Dm, Dm);
  k_cvt_wt<<<tg, tb, 0, stream>>>(Wv, wtv, Dm, Dm);
  k_cvt_wt<<<tg, tb, 0, stream>>>(W1, wt1, Dm, Dm);
  k_cvt_wt<<<tg, tb, 0, stream>>>(W2, wt2, Dm, Dm);

  dim3 gg(Dm / 128, Mtot / 128);
  k_gemm_wmma<<<gg, 256, 0, stream>>>(qb, wtq, bq, nullptr, nullptr, Qbm,
                                      Mtot, Dm, Dm, 0, 0.125f);   // 1/sqrt(64) folded
  k_gemm_wmma<<<gg, 256, 0, stream>>>(qb, wtk, bk, nullptr, nullptr, Kbm,
                                      Mtot, Dm, Dm, 0, 1.0f);
  k_gemm_wmma<<<gg, 256, 0, stream>>>(qb, wtv, bv, nullptr, nullptr, Vtm,
                                      Mtot, Dm, Dm, 1, 1.0f);     // transposed store

  k_attn<<<Bb * L / 16, 512, 0, stream>>>(Qbm, Kbm, Vtm, Wm, bm, attn);
  k_add_ln<<<Mtot / 8, 256, 0, stream>>>(attn, q, lng, lnb, xf, xb);

  k_gemm_wmma<<<gg, 256, 0, stream>>>(xb, wt1, b1, nullptr, nullptr, hb,
                                      Mtot, Dm, Dm, 2, 1.0f);     // GELU
  k_gemm_wmma<<<gg, 256, 0, stream>>>(hb, wt2, b2, xf, (float*)d_out, nullptr,
                                      Mtot, Dm, Dm, 3, 1.0f);     // + residual
  (void)in_sizes; (void)n_in; (void)out_size; (void)ws_size;
}
